// GCN_6588479832234
// MI455X (gfx1250) — matrix operand order repaired and checked by reading, not compiled, not run on hardware
//
#include <hip/hip_runtime.h>
#include <hip/hip_bf16.h>
#include <math.h>

// ---------------------------------------------------------------------------
// CDNA5 (gfx1250) GAT + TopK-pool GNN pipeline.
// GEMMs use v_wmma_f32_16x16x32_f16 (f16 in, f32 accumulate).
// Edge aggregation is L2-resident f32 atomic scatter (memory-roofline bound).
// ---------------------------------------------------------------------------

typedef __attribute__((ext_vector_type(16))) _Float16 v16h;
typedef __attribute__((ext_vector_type(8)))  _Float16 v8h;
typedef __attribute__((ext_vector_type(8)))  float    v8f;

union V16H { v16h v; v8h h[2]; };

// ---- ordered-float encoding for atomicMax on f32 --------------------------
__device__ __forceinline__ unsigned fenc(float f) {
  unsigned u = __float_as_uint(f);
  return (u & 0x80000000u) ? ~u : (u | 0x80000000u);
}
__device__ __forceinline__ float fdec(unsigned u) {
  u = (u & 0x80000000u) ? (u & 0x7fffffffu) : ~u;
  return __uint_as_float(u);
}
// fenc(-INF) == 0x007FFFFF
#define MENC_NEG_INF 0x007FFFFFu

// ---- small utility kernels ------------------------------------------------
__global__ void k_fill_f32(float* p, float v, int n) {
  int i = blockIdx.x * blockDim.x + threadIdx.x;
  if (i < n) p[i] = v;
}
__global__ void k_fill_u32(unsigned* p, unsigned v, int n) {
  int i = blockIdx.x * blockDim.x + threadIdx.x;
  if (i < n) p[i] = v;
}
__global__ void k_f32_to_f16(const float* __restrict__ s, _Float16* __restrict__ d, int n) {
  int i = blockIdx.x * blockDim.x + threadIdx.x;
  if (i < n) d[i] = (_Float16)s[i];
}
__global__ void k_bias_init(float* __restrict__ o, const float* __restrict__ b, int n128) {
  int i = blockIdx.x * blockDim.x + threadIdx.x;
  if (i < n128) o[i] = b[i & 127];
}
__global__ void k_relu(float* x, int n) {
  int i = blockIdx.x * blockDim.x + threadIdx.x;
  if (i < n) x[i] = fmaxf(x[i], 0.0f);
}
__global__ void k_edge_copy(const int* __restrict__ ei, int* __restrict__ src,
                            int* __restrict__ dst, int* __restrict__ val, int E) {
  int i = blockIdx.x * blockDim.x + threadIdx.x;
  if (i >= E) return;
  src[i] = ei[i];
  dst[i] = ei[E + i];
  val[i] = 1;
}

// ---- weight pre-swizzle into WMMA B-fragment layout -----------------------
// B(k,j) = W[j][k].  Tile t = kkIdx*8 + j0 (32 tiles of 32x16).  Within a
// tile, lane L half hh maps to k = kkIdx*32 + (L<16?0:16) + hh, j = j0*16 + (L&15).
__global__ void k_swizzle_w(const float* __restrict__ W, _Float16* __restrict__ ws) {
  int t = blockIdx.x * blockDim.x + threadIdx.x;
  if (t >= 128 * 128) return;
  int tile = t >> 9, rem = t & 511;
  int L = rem >> 4, hh = rem & 15;
  int kkIdx = tile >> 3, j0 = tile & 7;
  int k = (kkIdx << 5) + ((L < 16) ? 0 : 16) + hh;
  int j = (j0 << 4) + (L & 15);
  ws[t] = (_Float16)W[j * 128 + k];
}

// ---- h = x @ W^T via v_wmma_f32_16x16x32_f16 ------------------------------
// Block = 256 threads = 8 waves.  Wave w owns the 16x16 output tile at
// column tile w; block owns a 16-row strip.  K=128 -> 4 WMMA per tile.
__global__ void k_gemm_wmma(const _Float16* __restrict__ xh,
                            const _Float16* __restrict__ ws,
                            float* __restrict__ out, int Nn) {
  const int lane = threadIdx.x & 31;
  const int wv   = threadIdx.x >> 5;   // column tile 0..7
  const int row0 = blockIdx.x << 4;
  const int m    = lane & 15;
  const int hi   = lane >> 4;          // 0/1 half-wave
  v8f acc;
#pragma unroll
  for (int r = 0; r < 8; ++r) acc[r] = 0.0f;
#pragma unroll
  for (int kkIdx = 0; kkIdx < 4; ++kkIdx) {
    const int kk = kkIdx << 5;
    V16H a, b;
    // A fragment: lane holds row (row0+m), K in {kk+hi*16 .. +15}
    const _Float16* ap = xh + (size_t)(row0 + m) * 128 + kk + (hi << 4);
    a.h[0] = *(const v8h*)(ap);
    a.h[1] = *(const v8h*)(ap + 8);
    // B fragment: pre-swizzled, 32 contiguous bytes per lane
    const _Float16* bp = ws + (((kkIdx << 3) + wv) << 9) + (lane << 4);
    b.h[0] = *(const v8h*)(bp);
    b.h[1] = *(const v8h*)(bp + 8);
    acc = __builtin_amdgcn_wmma_f32_16x16x32_f16(false, a.v, false, b.v,
                                                 (short)0, acc, false, false);
  }
  // C/D layout: VGPR r -> M = r (lanes 0-15) / 8+r (lanes 16-31), N = lane&15
  float* o = out + (size_t)row0 * 128 + (wv << 4) + m;
#pragma unroll
  for (int r = 0; r < 8; ++r)
    o[(size_t)((hi ? 8 : 0) + r) * 128] = acc[r];
}

// ---- per-node attention projections: as=h.aS, ad=h.aD (wave per node) -----
__global__ void k_alpha(const float* __restrict__ h, const float* __restrict__ aS,
                        const float* __restrict__ aD, float* __restrict__ as_,
                        float* __restrict__ ad_, int Nn) {
  int wid = (blockIdx.x * blockDim.x + threadIdx.x) >> 5;
  int lane = threadIdx.x & 31;
  if (wid >= Nn) return;
  const float* hr = h + (size_t)wid * 128;
  float s = 0.f, d = 0.f;
#pragma unroll
  for (int f = lane; f < 128; f += 32) { float v = hr[f]; s += v * aS[f]; d += v * aD[f]; }
#pragma unroll
  for (int o = 16; o > 0; o >>= 1) { s += __shfl_xor(s, o, 32); d += __shfl_xor(d, o, 32); }
  if (lane == 0) { as_[wid] = s; ad_[wid] = d; }
}

// ---- edge pass A: logits + segment max (self loops are slots >= E) --------
__global__ void k_edge_logit(const int* __restrict__ src, const int* __restrict__ dst,
                             const int* __restrict__ val, const float* __restrict__ as_,
                             const float* __restrict__ ad_, float* __restrict__ ea,
                             unsigned* __restrict__ me, int E, int Nn) {
  int i = blockIdx.x * blockDim.x + threadIdx.x;
  if (i >= E + Nn) return;
  int s, d, v;
  if (i < E) { v = val[i]; s = src[i]; d = dst[i]; } else { s = d = i - E; v = 1; }
  float lg = -INFINITY;
  if (v) {
    float t = as_[s] + ad_[d];
    lg = t > 0.f ? t : 0.2f * t;                  // leaky_relu(., 0.2)
    atomicMax(&me[d], fenc(lg));
  }
  ea[i] = lg;
}

// ---- edge pass B: e = exp(logit - m[d]); den[d] += e ----------------------
__global__ void k_edge_exp(const int* __restrict__ src, const int* __restrict__ dst,
                           const int* __restrict__ val, const unsigned* __restrict__ me,
                           float* __restrict__ ea, float* __restrict__ den, int E, int Nn) {
  int i = blockIdx.x * blockDim.x + threadIdx.x;
  if (i >= E + Nn) return;
  int d, v;
  if (i < E) { v = val[i]; d = dst[i]; } else { d = i - E; v = 1; }
  if (v) {
    float e = expf(ea[i] - fdec(me[d]));
    atomicAdd(&den[d], e);
    ea[i] = e;
  } else {
    ea[i] = 0.f;
  }
}

// ---- edge pass C: out[d] += (e/den[d]) * h[s]  (wave per edge slot) -------
__global__ void k_edge_agg(const int* __restrict__ src, const int* __restrict__ dst,
                           const float* __restrict__ ea, const float* __restrict__ den,
                           const float* __restrict__ h, float* __restrict__ out,
                           int E, int Nn) {
  int gt = blockIdx.x * blockDim.x + threadIdx.x;
  int slot = gt >> 5, lane = gt & 31;
  if (slot >= E + Nn) return;
  float e = ea[slot];
  if (e == 0.f) return;
  int s, d;
  if (slot < E) { s = src[slot]; d = dst[slot]; } else { s = d = slot - E; }
  float w = e / den[d];
  const float* hs = h + (size_t)s * 128;
  float* od = out + (size_t)d * 128;
#pragma unroll
  for (int f = lane; f < 128; f += 32) atomicAdd(&od[f], w * hs[f]);
}

// ---- ||p|| ----------------------------------------------------------------
__global__ void k_vecnorm(const float* __restrict__ p, float* __restrict__ pn) {
  __shared__ float sm[128];
  int t = threadIdx.x;
  float v = p[t];
  sm[t] = v * v;
  __syncthreads();
  for (int s = 64; s > 0; s >>= 1) { if (t < s) sm[t] += sm[t + s]; __syncthreads(); }
  if (t == 0) pn[0] = sqrtf(sm[0]);
}

// ---- TopK pooling: one block per graph, LDS bitonic sort (descending) -----
__global__ void k_pool_topk(const float* __restrict__ xin, const float* __restrict__ p,
                            const float* __restrict__ pn, float* __restrict__ xout,
                            int* __restrict__ mapping, int n_in, int k) {
  __shared__ float ss[1024];
  __shared__ int   si[1024];
  const int g = blockIdx.x, tid = threadIdx.x, nP = blockDim.x;
  float sc = -INFINITY;
  if (tid < n_in) {
    const float* xr = xin + (size_t)(g * n_in + tid) * 128;
    float acc = 0.f;
    for (int f = 0; f < 128; ++f) acc += xr[f] * p[f];
    sc = acc / pn[0];
    mapping[g * n_in + tid] = -1;
  }
  ss[tid] = sc; si[tid] = tid;
  for (int size = 2; size <= nP; size <<= 1)
    for (int stride = size >> 1; stride > 0; stride >>= 1) {
      __syncthreads();
      int j = tid ^ stride;
      if (j > tid) {
        bool desc = ((tid & size) == 0);
        float a = ss[tid], b = ss[j];
        bool sw = desc ? (a < b) : (a > b);
        if (sw) { ss[tid] = b; ss[j] = a; int t = si[tid]; si[tid] = si[j]; si[j] = t; }
      }
    }
  __syncthreads();
  if (tid < k) mapping[g * n_in + si[tid]] = g * k + tid;
  __syncthreads();
  for (int j = tid; j < k * 128; j += nP) {
    int r = j >> 7, f = j & 127;
    xout[(size_t)(g * k + r) * 128 + f] =
        xin[(size_t)(g * n_in + si[r]) * 128 + f] * tanhf(ss[r]);
  }
}

// ---- filter_adj via mapping ----------------------------------------------
__global__ void k_edge_remap(int* src, int* dst, int* val, const int* __restrict__ mapping, int E) {
  int i = blockIdx.x * blockDim.x + threadIdx.x;
  if (i >= E) return;
  int v = val[i], ns = 0, nd = 0, nv = 0;
  if (v) {
    int a = mapping[src[i]], b = mapping[dst[i]];
    if (a >= 0 && b >= 0) { ns = a; nd = b; nv = 1; }
  }
  src[i] = ns; dst[i] = nd; val[i] = nv;
}

// ---- readout [gmp | gap] accumulated into z -------------------------------
__global__ void k_readout(const float* __restrict__ x, float* __restrict__ z, int k) {
  int g = blockIdx.x, f = threadIdx.x;   // 128 threads
  const float* xr = x + (size_t)g * k * 128 + f;
  float mx = -INFINITY, sm = 0.f;
  for (int r = 0; r < k; ++r) { float v = xr[(size_t)r * 128]; mx = fmaxf(mx, v); sm += v; }
  z[g * 256 + f]       += mx;
  z[g * 256 + 128 + f] += sm / (float)k;
}

// ---- MLP head + log_softmax: one block per graph --------------------------
__global__ void k_mlp(const float* __restrict__ z,
                      const float* __restrict__ lw1, const float* __restrict__ lb1,
                      const float* __restrict__ lw2, const float* __restrict__ lb2,
                      const float* __restrict__ lw3, const float* __restrict__ lb3,
                      float* __restrict__ out) {
  __shared__ float zr[256], z1[128], z2[64], lg[10], red[2];
  int g = blockIdx.x, t = threadIdx.x;   // 256 threads
  zr[t] = z[g * 256 + t];
  __syncthreads();
  if (t < 128) {
    float a = lb1[t]; const float* w = lw1 + t * 256;
    for (int c = 0; c < 256; ++c) a += zr[c] * w[c];
    z1[t] = fmaxf(a, 0.f);
  }
  __syncthreads();
  if (t < 64) {
    float a = lb2[t]; const float* w = lw2 + t * 128;
    for (int c = 0; c < 128; ++c) a += z1[c] * w[c];
    z2[t] = fmaxf(a, 0.f);
  }
  __syncthreads();
  if (t < 10) {
    float a = lb3[t]; const float* w = lw3 + t * 64;
    for (int c = 0; c < 64; ++c) a += z2[c] * w[c];
    lg[t] = a;
  }
  __syncthreads();
  if (t == 0) {
    float m = lg[0];
    for (int c = 1; c < 10; ++c) m = fmaxf(m, lg[c]);
    float s = 0.f;
    for (int c = 0; c < 10; ++c) s += expf(lg[c] - m);
    red[0] = m; red[1] = logf(s);
  }
  __syncthreads();
  if (t < 10) out[g * 10 + t] = lg[t] - red[0] - red[1];
}

// ---------------------------------------------------------------------------
extern "C" void kernel_launch(void* const* d_in, const int* in_sizes, int n_in,
                              void* d_out, int out_size, void* d_ws, size_t ws_size,
                              hipStream_t stream) {
  (void)in_sizes; (void)n_in; (void)out_size; (void)ws_size;
  const float* x  = (const float*)d_in[0];
  const int*   ei = (const int*)d_in[1];
  const float* Wl[3]  = {(const float*)d_in[2],  (const float*)d_in[6],  (const float*)d_in[10]};
  const float* aSl[3] = {(const float*)d_in[3],  (const float*)d_in[7],  (const float*)d_in[11]};
  const float* aDl[3] = {(const float*)d_in[4],  (const float*)d_in[8],  (const float*)d_in[12]};
  const float* bl[3]  = {(const float*)d_in[5],  (const float*)d_in[9],  (const float*)d_in[13]};
  const float* pl[3]  = {(const float*)d_in[14], (const float*)d_in[15], (const float*)d_in[15]};
  const float* lw1 = (const float*)d_in[16]; const float* lb1 = (const float*)d_in[17];
  const float* lw2 = (const float*)d_in[18]; const float* lb2 = (const float*)d_in[19];
  const float* lw3 = (const float*)d_in[20]; const float* lb3 = (const float*)d_in[21];
  float* out = (float*)d_out;

  const int E = 800000, G = 50, N0 = 50000;
  const int Nn[3]   = {50000, 30000, 18000};  // nodes entering each GAT layer
  const int nin[3]  = {1000, 600, 360};       // nodes/graph entering each pool
  const int kk[3]   = {600, 360, 216};        // nodes/graph kept by each pool
  const int np2[3]  = {1024, 1024, 512};      // bitonic sort sizes

  char* base = (char*)d_ws;
  size_t off = 0;
  auto alloc = [&](size_t bytes) -> void* {
    void* p = base + off;
    off = (off + bytes + 255) & ~(size_t)255;
    return p;
  };
  float*     F0  = (float*)alloc((size_t)N0 * 128 * 4);
  float*     F1  = (float*)alloc((size_t)N0 * 128 * 4);
  _Float16*  XH  = (_Float16*)alloc((size_t)N0 * 128 * 2);
  _Float16*  WS  = (_Float16*)alloc(128 * 128 * 2);
  float*     AS  = (float*)alloc((size_t)N0 * 4);
  float*     AD  = (float*)alloc((size_t)N0 * 4);
  float*     DEN = (float*)alloc((size_t)N0 * 4);
  unsigned*  ME  = (unsigned*)alloc((size_t)N0 * 4);
  float*     EA  = (float*)alloc((size_t)(E + N0) * 4);
  int*       SRC = (int*)alloc((size_t)E * 4);
  int*       DST = (int*)alloc((size_t)E * 4);
  int*       VAL = (int*)alloc((size_t)E * 4);
  int*       MAP = (int*)alloc((size_t)N0 * 4);
  float*     PN  = (float*)alloc(256);
  float*     Z   = (float*)alloc((size_t)G * 256 * 4);

  auto cdiv = [](int a, int b) { return (a + b - 1) / b; };

  k_edge_copy<<<cdiv(E, 256), 256, 0, stream>>>(ei, SRC, DST, VAL, E);
  k_fill_f32<<<cdiv(G * 256, 256), 256, 0, stream>>>(Z, 0.f, G * 256);

  for (int l = 0; l < 3; ++l) {
    const int N = Nn[l], EN = E + N;
    const float* cur = (l == 0) ? x : F0;
    // 1) f16 copies of activations + weight pre-swizzle
    k_f32_to_f16<<<cdiv(N * 128, 256), 256, 0, stream>>>(cur, XH, N * 128);
    k_swizzle_w<<<64, 256, 0, stream>>>(Wl[l], WS);
    // 2) h = x @ W^T  (WMMA)
    k_gemm_wmma<<<N / 16, 256, 0, stream>>>(XH, WS, F0, N);
    // 3) attention projections
    k_alpha<<<cdiv(N, 8), 256, 0, stream>>>(F0, aSl[l], aDl[l], AS, AD, N);
    // 4) softmax over incoming edges (with self loops)
    k_fill_u32<<<cdiv(N, 256), 256, 0, stream>>>(ME, MENC_NEG_INF, N);
    k_fill_f32<<<cdiv(N, 256), 256, 0, stream>>>(DEN, 0.f, N);
    k_bias_init<<<cdiv(N * 128, 256), 256, 0, stream>>>(F1, bl[l], N * 128);
    k_edge_logit<<<cdiv(EN, 256), 256, 0, stream>>>(SRC, DST, VAL, AS, AD, EA, ME, E, N);
    k_edge_exp<<<cdiv(EN, 256), 256, 0, stream>>>(SRC, DST, VAL, ME, EA, DEN, E, N);
    k_edge_agg<<<cdiv(EN * 32, 256), 256, 0, stream>>>(SRC, DST, EA, DEN, F0, F1, E, N);
    k_relu<<<cdiv(N * 128, 256), 256, 0, stream>>>(F1, N * 128);
    // 5) TopK pool (F1 -> F0) + edge filtering + readout
    k_vecnorm<<<1, 128, 0, stream>>>(pl[l], PN);
    k_pool_topk<<<G, np2[l], 0, stream>>>(F1, pl[l], PN, F0, MAP, nin[l], kk[l]);
    k_edge_remap<<<cdiv(E, 256), 256, 0, stream>>>(SRC, DST, VAL, MAP, E);
    k_readout<<<G, 128, 0, stream>>>(F0, Z, kk[l]);
  }
  // 6) MLP head + log_softmax
  k_mlp<<<G, 256, 0, stream>>>(Z, lw1, lb1, lw2, lb2, lw3, lb3, out);
}
